// MultiHead_57088705298785
// MI455X (gfx1250) — matrix operand order repaired and checked by reading, not compiled, not run on hardware
//
#include <hip/hip_runtime.h>
#include <hip/hip_bf16.h>

typedef __attribute__((ext_vector_type(16))) _Float16 v16h;
typedef __attribute__((ext_vector_type(8)))  _Float16 v8h;
typedef __attribute__((ext_vector_type(8)))  float    v8f;
typedef __attribute__((ext_vector_type(4)))  float    v4f;

constexpr int B_ = 256, T_ = 256, D_ = 384, H_ = 6, HS_ = 64;
constexpr int PLD = 48;  // LDS row stride for P tiles (multiple of 8 halves -> 16B aligned rows)

__device__ __forceinline__ v16h wmma_pack(v8h lo, v8h hi) {
  v16h a;
#pragma unroll
  for (int i = 0; i < 8; ++i) { a[i] = lo[i]; a[8 + i] = hi[i]; }
  return a;
}

// A fragment (16x32 f16) from row-major f16 matrix at (0,0)=base, leading dim lda (halves).
// ISA layout: lanes 0-15 hold K 0..7 / 16..23, lanes 16-31 hold K 8..15 / 24..31, M = lane%16.
__device__ __forceinline__ v16h load_a_f16(const _Float16* base, int lda, int lane) {
  const int row = lane & 15;
  const int kb  = (lane & 16) ? 8 : 0;
  const _Float16* p = base + (size_t)row * lda + kb;
  v8h lo = *(const v8h*)(p);
  v8h hi = *(const v8h*)(p + 16);
  return wmma_pack(lo, hi);
}

// A fragment from row-major fp32 source, converted to f16.
__device__ __forceinline__ v16h load_a_f32(const float* base, int lda, int lane) {
  const int row = lane & 15;
  const int kb  = (lane & 16) ? 8 : 0;
  const float* p = base + (size_t)row * lda + kb;
  v4f f0 = *(const v4f*)(p);
  v4f f1 = *(const v4f*)(p + 4);
  v4f f2 = *(const v4f*)(p + 16);
  v4f f3 = *(const v4f*)(p + 20);
  v16h a;
#pragma unroll
  for (int i = 0; i < 4; ++i) {
    a[i]      = (_Float16)f0[i];
    a[4 + i]  = (_Float16)f1[i];
    a[8 + i]  = (_Float16)f2[i];
    a[12 + i] = (_Float16)f3[i];
  }
  return a;
}

// B fragment (32x16, KxN) from "N-major" storage Bt where B[k][n] = Bt[n*ldb + k].
// ISA layout: N = lane%16; lanes 0-15 hold K 0..7 / 16..23, lanes 16-31 hold K 8..15 / 24..31.
__device__ __forceinline__ v16h load_b_t_f16(const _Float16* base, int ldb, int lane) {
  const int n  = lane & 15;
  const int kb = (lane & 16) ? 8 : 0;
  const _Float16* p = base + (size_t)n * ldb + kb;
  v8h lo = *(const v8h*)(p);
  v8h hi = *(const v8h*)(p + 16);
  return wmma_pack(lo, hi);
}

__device__ __forceinline__ v8f wmma_f16(v16h a, v16h b, v8f c) {
  return __builtin_amdgcn_wmma_f32_16x16x32_f16(false, a, false, b, (short)0, c, false, false);
}

// reductions across the 16-lane halves (rows of the C layout live per half-wave)
__device__ __forceinline__ float halfmax(float v) {
#pragma unroll
  for (int m = 1; m <= 8; m <<= 1) v = fmaxf(v, __shfl_xor(v, m, 32));
  return v;
}
__device__ __forceinline__ float halfsum(float v) {
#pragma unroll
  for (int m = 1; m <= 8; m <<= 1) v += __shfl_xor(v, m, 32);
  return v;
}

// ---------------- Kernel 0: transpose + fp32->f16 weights ----------------
__global__ void prep_weights(const float* __restrict__ Wqkv, const float* __restrict__ Wout,
                             _Float16* __restrict__ WtQkv, _Float16* __restrict__ WtOut) {
  const int i  = blockIdx.x * blockDim.x + threadIdx.x;
  const int n1 = H_ * D_ * 3 * HS_;
  const int n2 = D_ * D_;
  if (i < n1) {
    const int h = i / (D_ * 3 * HS_);
    const int r = i % (D_ * 3 * HS_);
    const int e = r / D_;
    const int d = r % D_;
    WtQkv[i] = (_Float16)Wqkv[((size_t)h * D_ + d) * (3 * HS_) + e];
  } else if (i < n1 + n2) {
    const int j  = i - n1;
    const int no = j / D_;
    const int k  = j % D_;
    WtOut[j] = (_Float16)Wout[(size_t)k * D_ + no];
  }
}

// ---------------- Kernel 1: QKV projection ----------------
// Streaming GEMM: A and B fragments are loaded inside the k-loop (no cross-tile
// register hoisting -> no spills). Each wave sweeps the 12 N-tiles of one M-tile
// consecutively so the redundant A loads hit L0.
__global__ void __launch_bounds__(256) qkv_proj(const float* __restrict__ x,
                                                const _Float16* __restrict__ WtQkv,
                                                _Float16* __restrict__ Q,
                                                _Float16* __restrict__ K,
                                                _Float16* __restrict__ Vt) {
  const int bh = blockIdx.x;
  const int b = bh / H_, h = bh % H_;
  const int wave = threadIdx.x >> 5, lane = threadIdx.x & 31;
  const int hl8 = (lane & 16) ? 8 : 0;
  const int cl  = lane & 15;
  const float*    xb = x + (size_t)b * T_ * D_;
  const _Float16* Wh = WtQkv + (size_t)h * (3 * HS_) * D_;   // [e][d]
  _Float16* Qbh  = Q  + (size_t)bh * T_ * HS_;
  _Float16* Kbh  = K  + (size_t)bh * T_ * HS_;
  _Float16* Vtbh = Vt + (size_t)bh * HS_ * T_;               // [e][t]

#pragma unroll 1
  for (int mi = 0; mi < 2; ++mi) {
    const int mt   = wave * 2 + mi;       // 16 M-tiles over 8 waves
    const int row0 = mt * 16;
#pragma unroll 1
    for (int nt = 0; nt < 12; ++nt) {
      v8f c = {};
#pragma unroll 4
      for (int k = 0; k < D_; k += 32) {
        v16h a  = load_a_f32(xb + (size_t)row0 * D_ + k, D_, lane);
        v16h bb = load_b_t_f16(Wh + (size_t)(nt * 16) * D_ + k, D_, lane);
        c = wmma_f16(a, bb, c);
      }
      const int col0 = nt * 16;
      if (col0 < HS_) {                   // Q (fold softmax scale 1/sqrt(64))
#pragma unroll
        for (int r = 0; r < 8; ++r)
          Qbh[(size_t)(row0 + hl8 + r) * HS_ + (col0 + cl)] = (_Float16)(c[r] * 0.125f);
      } else if (col0 < 2 * HS_) {        // K
#pragma unroll
        for (int r = 0; r < 8; ++r)
          Kbh[(size_t)(row0 + hl8 + r) * HS_ + (col0 - HS_ + cl)] = (_Float16)c[r];
      } else {                            // V stored transposed [e][t]
#pragma unroll
        for (int r = 0; r < 8; ++r)
          Vtbh[(size_t)(col0 - 2 * HS_ + cl) * T_ + (row0 + hl8 + r)] = (_Float16)c[r];
      }
    }
  }
}

// ---------------- Kernel 2: flash attention per (b,h) ----------------
__global__ void __launch_bounds__(256) attn(const _Float16* __restrict__ Q,
                                            const _Float16* __restrict__ K,
                                            const _Float16* __restrict__ Vt,
                                            _Float16* __restrict__ O) {
  __shared__ __attribute__((aligned(16))) _Float16 Plds[8][32][PLD];
  const int bh = blockIdx.x;
  const int b = bh / H_, h = bh % H_;
  const int wave = threadIdx.x >> 5, lane = threadIdx.x & 31;
  const int hl8 = (lane & 16) ? 8 : 0;
  const int cl  = lane & 15;
  const int qrow0 = wave * 32;                 // 32 query rows per wave
  const _Float16* Qbh  = Q  + (size_t)bh * T_ * HS_;
  const _Float16* Kbh  = K  + (size_t)bh * T_ * HS_;
  const _Float16* Vtbh = Vt + (size_t)bh * HS_ * T_;

  v16h aq[2][2];
#pragma unroll
  for (int mt = 0; mt < 2; ++mt)
#pragma unroll
    for (int ks = 0; ks < 2; ++ks)
      aq[mt][ks] = load_a_f16(Qbh + (size_t)(qrow0 + mt * 16) * HS_ + ks * 32, HS_, lane);

  const v8f vzero = {};
  float mrow[2][8], lrow[2][8];
  v8f o[2][4];
#pragma unroll
  for (int mt = 0; mt < 2; ++mt) {
#pragma unroll
    for (int r = 0; r < 8; ++r) { mrow[mt][r] = -3.0e38f; lrow[mt][r] = 0.0f; }
#pragma unroll
    for (int ne = 0; ne < 4; ++ne) o[mt][ne] = vzero;
  }

#pragma unroll 1
  for (int sb = 0; sb <= wave; ++sb) {         // causal: skip upper-triangular blocks
    const int s0 = sb * 32;
    v8f st[2][2];
#pragma unroll
    for (int mt = 0; mt < 2; ++mt)
#pragma unroll
      for (int nt = 0; nt < 2; ++nt) {
        v8f c = vzero;
#pragma unroll
        for (int ks = 0; ks < 2; ++ks) {
          v16h bk = load_b_t_f16(Kbh + (size_t)(s0 + nt * 16) * HS_ + ks * 32, HS_, lane);
          c = wmma_f16(aq[mt][ks], bk, c);     // S = (Q*scale) K^T
        }
        st[mt][nt] = c;
      }
    const bool edge = (sb == wave);
#pragma unroll
    for (int mt = 0; mt < 2; ++mt) {
#pragma unroll
      for (int r = 0; r < 8; ++r) {
        const int row = qrow0 + mt * 16 + hl8 + r;
        float s0v = st[mt][0][r];
        float s1v = st[mt][1][r];
        if (edge) {
          if (s0 + cl      > row) s0v = -1.0e30f;
          if (s0 + 16 + cl > row) s1v = -1.0e30f;
        }
        float mx = halfmax(fmaxf(s0v, s1v));
        const float mnew  = fmaxf(mrow[mt][r], mx);
        const float alpha = __expf(mrow[mt][r] - mnew);
        const float p0 = __expf(s0v - mnew);
        const float p1 = __expf(s1v - mnew);
        lrow[mt][r] = lrow[mt][r] * alpha + halfsum(p0 + p1);
        mrow[mt][r] = mnew;
#pragma unroll
        for (int ne = 0; ne < 4; ++ne) o[mt][ne][r] *= alpha;
        Plds[wave][mt * 16 + hl8 + r][cl]      = (_Float16)p0;
        Plds[wave][mt * 16 + hl8 + r][16 + cl] = (_Float16)p1;
      }
    }
    // wave-private LDS round trip (C-layout -> A-layout); no block barrier needed.
    asm volatile("s_wait_dscnt 0" ::: "memory");
#pragma unroll
    for (int mt = 0; mt < 2; ++mt) {
      v16h ap = load_a_f16(&Plds[wave][mt * 16][0], PLD, lane);
#pragma unroll
      for (int ne = 0; ne < 4; ++ne) {
        v16h bv = load_b_t_f16(Vtbh + (size_t)(ne * 16) * T_ + s0, T_, lane);
        o[mt][ne] = wmma_f16(ap, bv, o[mt][ne]);   // O += P V
      }
    }
  }

  _Float16* Ob = O + ((size_t)b * T_ + qrow0) * (H_ * HS_) + h * HS_;
#pragma unroll
  for (int mt = 0; mt < 2; ++mt)
#pragma unroll
    for (int r = 0; r < 8; ++r) {
      const float inv = 1.0f / lrow[mt][r];
#pragma unroll
      for (int ne = 0; ne < 4; ++ne)
        Ob[(size_t)(mt * 16 + hl8 + r) * (H_ * HS_) + ne * 16 + cl] =
            (_Float16)(o[mt][ne][r] * inv);
    }
}

// ---------------- Kernel 3: output projection + bias ----------------
__global__ void __launch_bounds__(256) out_proj(const _Float16* __restrict__ O,
                                                const _Float16* __restrict__ WtOut,
                                                const float* __restrict__ bias,
                                                float* __restrict__ out) {
  const int rowblk = blockIdx.x;               // 16-row panel of [B*T, D]
  const int wave = threadIdx.x >> 5, lane = threadIdx.x & 31;
  const int hl8 = (lane & 16) ? 8 : 0;
  const int cl  = lane & 15;
  const _Float16* Arow = O + (size_t)rowblk * 16 * D_;

#pragma unroll 1
  for (int nt = wave; nt < D_ / 16; nt += 8) {
    v8f c = {};
#pragma unroll 4
    for (int k = 0; k < D_; k += 32) {
      v16h a  = load_a_f16(Arow + k, D_, lane);
      v16h bb = load_b_t_f16(WtOut + (size_t)(nt * 16) * D_ + k, D_, lane);
      c = wmma_f16(a, bb, c);
    }
    const int col = nt * 16 + cl;
    const float bv = bias[col];
#pragma unroll
    for (int r = 0; r < 8; ++r)
      out[((size_t)rowblk * 16 + hl8 + r) * D_ + col] = c[r] + bv;
  }
}

extern "C" void kernel_launch(void* const* d_in, const int* in_sizes, int n_in,
                              void* d_out, int out_size, void* d_ws, size_t ws_size,
                              hipStream_t stream) {
  const float* x    = (const float*)d_in[0];
  const float* Wqkv = (const float*)d_in[1];
  const float* Wout = (const float*)d_in[2];
  const float* bias = (const float*)d_in[3];
  float* out = (float*)d_out;

  // Workspace layout (f16), ~203 MB total:
  _Float16* WtQkv = (_Float16*)d_ws;                         // H*192*384
  _Float16* WtOut = WtQkv + (size_t)H_ * (3 * HS_) * D_;     // 384*384
  _Float16* Q     = WtOut + (size_t)D_ * D_;                 // B*H*T*HS each:
  const size_t NQ = (size_t)B_ * H_ * T_ * HS_;
  _Float16* K  = Q  + NQ;
  _Float16* Vt = K  + NQ;
  _Float16* O  = Vt + NQ;                                    // [B*T][H*HS]

  const int total = H_ * D_ * 3 * HS_ + D_ * D_;
  prep_weights<<<(total + 255) / 256, 256, 0, stream>>>(Wqkv, Wout, WtQkv, WtOut);
  qkv_proj<<<B_ * H_, 256, 0, stream>>>(x, WtQkv, Q, K, Vt);
  attn<<<B_ * H_, 256, 0, stream>>>(Q, K, Vt, O);
  out_proj<<<B_ * T_ / 16, 256, 0, stream>>>(O, WtOut, bias, out);
}
